// MultiHeadSimilarityAttention_60129542144353
// MI455X (gfx1250) — matrix operand order repaired and checked by reading, not compiled, not run on hardware
//
#include <hip/hip_runtime.h>

#define B_    4
#define N_    2048
#define DIN   256
#define DM    512
#define H_    8
#define DH    64
#define TOPK_ 32

typedef float  v8f   __attribute__((ext_vector_type(8)));
typedef __bf16 v16bf __attribute__((ext_vector_type(16)));
typedef __bf16 v8bf  __attribute__((ext_vector_type(8)));

union V16U { v16bf v; v8bf h[2]; };

static __device__ inline __bf16 f2bf(float f) {
  unsigned u = __builtin_bit_cast(unsigned, f);
  unsigned r = u + 0x7FFFu + ((u >> 16) & 1u);   // round-to-nearest-even
  unsigned short s = (unsigned short)(r >> 16);
  return __builtin_bit_cast(__bf16, s);
}

static __device__ inline v8f wmma_bf16(v16bf a, v16bf b, v8f c) {
  // D(16x16,f32) = A(16x32,bf16) x B(32x16,bf16) + C
  return __builtin_amdgcn_wmma_f32_16x16x32_bf16(false, a, false, b,
                                                 (short)0, c, false, false);
}

// Async global -> LDS copy (16B per lane), tracked with ASYNCcnt.
static __device__ inline void async_load_b128(unsigned lds_off, const void* gaddr) {
  asm volatile("global_load_async_to_lds_b128 %0, %1, off"
               :: "v"(lds_off), "v"((unsigned long long)(size_t)gaddr)
               : "memory");
}
static __device__ inline void wait_async0() {
  asm volatile("s_wait_asynccnt 0" ::: "memory");
}

// ---------------------------------------------------------------------------
// Generic tiled GEMM:  out = A(MxK,f32) * W(KxN,f32) + bias(N)
// 128x64 tile per block (8 waves x 16 rows, 4 n-chunks of 16).
// ---------------------------------------------------------------------------
__global__ __launch_bounds__(256) void gemm_kernel(
    const float* __restrict__ A, const float* __restrict__ W,
    const float* __restrict__ bias, __bf16* __restrict__ outB,
    float* __restrict__ outF, int M, int K, int Nn, int storeF32)
{
  __shared__ __attribute__((aligned(32))) __bf16 Abuf[128][32];
  __shared__ __attribute__((aligned(32))) __bf16 WT[64][32];   // transposed W tile

  int t  = threadIdx.x;
  int w  = t >> 5, L = t & 31, r = L & 15, hi = L >> 4;
  int rowBase = blockIdx.x * 128;
  int nBase   = blockIdx.y * 64;

  v8f acc[4];
  for (int c = 0; c < 4; ++c)
    for (int i = 0; i < 8; ++i) acc[c][i] = 0.f;

  for (int kb = 0; kb < K; kb += 32) {
    __syncthreads();
    // A tile 128x32 (f32 -> bf16), coalesced; 32-bit offsets off a hoisted base
    {
      const float* Ab = A + (size_t)rowBase * K + kb;
      for (int i = 0; i < 16; ++i) {
        int lin = i * 256 + t;
        int row = lin >> 5, col = lin & 31;
        Abuf[row][col] = f2bf(Ab[row * K + col]);
      }
    }
    // W tile 32x64 stored transposed
    {
      const float* Wb = W + (size_t)kb * Nn + nBase;
      for (int i = 0; i < 8; ++i) {
        int lin = i * 256 + t;
        int kk = lin >> 6, nn = lin & 63;
        WT[nn][kk] = f2bf(Wb[kk * Nn + nn]);
      }
    }
    __syncthreads();

    // A fragment: row = w*16 + r ; K packing (e&7) + 16*(e>>3) + 8*hi
    V16U af;
    af.h[0] = *(const v8bf*)(&Abuf[w * 16 + r][8 * hi]);
    af.h[1] = *(const v8bf*)(&Abuf[w * 16 + r][16 + 8 * hi]);
    for (int c = 0; c < 4; ++c) {
      // B fragment: b[e] = WT[16c+r][(e&7) + 16*(e>>3) + 8*hi]
      V16U bf;
      bf.h[0] = *(const v8bf*)(&WT[16 * c + r][8 * hi]);
      bf.h[1] = *(const v8bf*)(&WT[16 * c + r][16 + 8 * hi]);
      acc[c] = wmma_bf16(af.v, bf.v, acc[c]);
    }
  }

  for (int c = 0; c < 4; ++c) {
    float bvv = bias[nBase + 16 * c + r];
    for (int v = 0; v < 8; ++v) {
      int row = rowBase + w * 16 + v + 8 * hi;   // C layout: row = VGPR + 8*hi
      int col = nBase + 16 * c + r;              //           col = lane%16
      float val = acc[c][v] + bvv;
      if (storeF32) outF[(size_t)row * Nn + col] = val;
      else          outB[(size_t)row * Nn + col] = f2bf(val);
    }
  }
}

// ---------------------------------------------------------------------------
// Top-k(32) of clamped similarity per row (diag excluded), row-normalized.
// One block per (b, n) row; iterative argmax (ties -> lowest index).
// ---------------------------------------------------------------------------
__global__ __launch_bounds__(256) void topk_kernel(
    const float* __restrict__ sim, int* __restrict__ biasIdx,
    float* __restrict__ biasVal)
{
  __shared__ float simArr[N_];
  __shared__ float redV[256];
  __shared__ int   redI[256];
  __shared__ float topVal[TOPK_];
  __shared__ int   topIdx[TOPK_];
  __shared__ float rowsumSh;

  int t  = threadIdx.x;
  int gr = blockIdx.x;            // 0 .. B*N-1
  int n  = gr % N_;
  const float* srow = sim + (size_t)gr * N_;

  for (int i = 0; i < N_ / 256; ++i) {
    int col = t + i * 256;
    float v = srow[col];
    v = v > 0.f ? v : 0.f;
    if (col == n) v = -INFINITY;
    simArr[col] = v;
  }
  __syncthreads();

  for (int k = 0; k < TOPK_; ++k) {
    float bv = -INFINITY; int bi = 0x7fffffff;
    for (int i = 0; i < N_ / 256; ++i) {
      int col = t + i * 256;
      float s = simArr[col];
      if (s > bv || (s == bv && col < bi)) { bv = s; bi = col; }
    }
    redV[t] = bv; redI[t] = bi;
    __syncthreads();
    for (int s = 128; s > 0; s >>= 1) {
      if (t < s) {
        float ov = redV[t + s]; int oi = redI[t + s];
        if (ov > redV[t] || (ov == redV[t] && oi < redI[t])) {
          redV[t] = ov; redI[t] = oi;
        }
      }
      __syncthreads();
    }
    if (t == 0) {
      topIdx[k] = redI[0];
      topVal[k] = redV[0];
      simArr[redI[0]] = -INFINITY;
    }
    __syncthreads();
  }

  if (t == 0) {
    float s = 0.f;
    for (int k = 0; k < TOPK_; ++k) s += topVal[k];
    rowsumSh = s > 1e-8f ? s : 1e-8f;
  }
  __syncthreads();
  if (t < TOPK_) {
    biasIdx[(size_t)gr * TOPK_ + t] = topIdx[t];
    biasVal[(size_t)gr * TOPK_ + t] = topVal[t] / rowsumSh;
  }
}

// ---------------------------------------------------------------------------
// Flash attention with sparse bias. Block = (b,h) x 128 rows, 8 waves.
// K tile staged via GLOBAL_LOAD_ASYNC_TO_LDS_B128 (ASYNCcnt); V tile
// transposed through VGPRs. S and PV on v_wmma_f32_16x16x32_bf16.
// ---------------------------------------------------------------------------
__global__ __launch_bounds__(256) void attn_kernel(
    const __bf16* __restrict__ Qb, const __bf16* __restrict__ Kb,
    const __bf16* __restrict__ Vb, const int* __restrict__ biasIdx,
    const float* __restrict__ biasVal, const float* __restrict__ tauRaw,
    float* __restrict__ attnO)
{
  __shared__ __attribute__((aligned(32))) __bf16 Kt[32][64];   // key tile
  __shared__ __attribute__((aligned(32))) __bf16 VtT[64][32];  // V tile, transposed
  __shared__ float Sbuf[128][33];                              // padded f32 logits
  __shared__ __attribute__((aligned(16))) __bf16 Pbuf[128][32];
  __shared__ float mRun[128], lRun[128], scaleArr[128];

  int t  = threadIdx.x;
  int w  = t >> 5, L = t & 31, r = L & 15, hi = L >> 4;
  int bh = blockIdx.y;
  int b  = bh / H_, h = bh % H_;
  int rowBase = blockIdx.x * 128;

  float x0  = tauRaw[0];
  float tau = (x0 > 20.f) ? x0 : log1pf(expf(x0));

  if (t < 128) { mRun[t] = -INFINITY; lRun[t] = 0.f; }

  // Q fragments for this wave's 16 rows (k = 0..63 in two 32-chunks)
  const __bf16* qrow = Qb + ((size_t)(b * N_ + rowBase + w * 16 + r)) * DM + h * DH;
  V16U q0, q1;
  q0.h[0] = *(const v8bf*)(qrow + 8 * hi);
  q0.h[1] = *(const v8bf*)(qrow + 16 + 8 * hi);
  q1.h[0] = *(const v8bf*)(qrow + 32 + 8 * hi);
  q1.h[1] = *(const v8bf*)(qrow + 48 + 8 * hi);

  // Sparse bias pairs in registers: row lr_b = t/2, 16 pairs per thread
  int lr_b = t >> 1;
  int bIdxReg[16]; float bValReg[16];
  {
    size_t base = ((size_t)(b * N_ + rowBase + lr_b)) * TOPK_ + (t & 1) * 16;
    for (int p = 0; p < 16; ++p) {
      bIdxReg[p] = biasIdx[base + p];
      bValReg[p] = tau * biasVal[base + p];
    }
  }

  v8f O[4];
  for (int c = 0; c < 4; ++c)
    for (int i = 0; i < 8; ++i) O[c][i] = 0.f;

  // Per-thread tile-load geometry: 32 rows x 64 cols, 16B per thread
  int kr = t >> 3, c0 = (t & 7) * 8;
  const __bf16* Khead = Kb + (size_t)(b * N_) * DM + h * DH;   // + (j0+kr)*DM + c0
  const __bf16* Vhead = Vb + (size_t)(b * N_) * DM + h * DH;
  unsigned ktOfs = (unsigned)(size_t)(void*)&Kt[kr][c0];       // LDS byte offset

  for (int j0 = 0; j0 < N_; j0 += 32) {
    __syncthreads();
    int go = (j0 + kr) * DM + c0;
    // K tile: async global -> LDS (no VGPR round trip)
    async_load_b128(ktOfs, Khead + go);
    // V tile through VGPRs (needs transpose into LDS)
    {
      v8bf vv = *(const v8bf*)(Vhead + go);
      for (int i = 0; i < 8; ++i) VtT[c0 + i][kr] = vv[i];
    }
    // Prefetch next tile's lines into cache while we compute this one
    if (j0 + 32 < N_) {
      __builtin_prefetch(Khead + go + 32 * DM, 0, 0);
      __builtin_prefetch(Vhead + go + 32 * DM, 0, 0);
    }
    wait_async0();
    __syncthreads();

    // S = Q * K^T (two 16-col halves, k = 64 in two chunks)
    for (int tt = 0; tt < 2; ++tt) {
      v8f sa; for (int i = 0; i < 8; ++i) sa[i] = 0.f;
      V16U b0, b1;
      b0.h[0] = *(const v8bf*)(&Kt[16 * tt + r][8 * hi]);
      b0.h[1] = *(const v8bf*)(&Kt[16 * tt + r][16 + 8 * hi]);
      sa = wmma_bf16(q0.v, b0.v, sa);
      b1.h[0] = *(const v8bf*)(&Kt[16 * tt + r][32 + 8 * hi]);
      b1.h[1] = *(const v8bf*)(&Kt[16 * tt + r][48 + 8 * hi]);
      sa = wmma_bf16(q1.v, b1.v, sa);
      for (int v = 0; v < 8; ++v)
        Sbuf[w * 16 + v + 8 * hi][16 * tt + r] = sa[v] * 0.125f; // 1/sqrt(64)
    }
    __syncthreads();

    // Sparse bias scatter into this key tile
    for (int p = 0; p < 16; ++p) {
      unsigned d = (unsigned)(bIdxReg[p] - j0);
      if (d < 32u) Sbuf[lr_b][d] += bValReg[p];
    }
    __syncthreads();

    // Online softmax row pass (one thread per row)
    if (t < 128) {
      float mOld = mRun[t];
      float mt = -INFINITY;
      for (int c = 0; c < 32; ++c) { float s = Sbuf[t][c]; mt = s > mt ? s : mt; }
      float mNew = mOld > mt ? mOld : mt;
      float sc = __expf(mOld - mNew);     // 0 on first tile (mOld = -inf)
      float sum = 0.f;
      for (int c = 0; c < 32; ++c) {
        float p = __expf(Sbuf[t][c] - mNew);
        Pbuf[t][c] = f2bf(p);
        sum += p;
      }
      lRun[t] = lRun[t] * sc + sum;
      mRun[t] = mNew;
      scaleArr[t] = sc;
    }
    __syncthreads();

    // Rescale O and accumulate P * V
    float scv[8];
    for (int v = 0; v < 8; ++v) scv[v] = scaleArr[w * 16 + v + 8 * hi];
    for (int c = 0; c < 4; ++c)
      for (int v = 0; v < 8; ++v) O[c][v] *= scv[v];
    V16U pf;
    pf.h[0] = *(const v8bf*)(&Pbuf[w * 16 + r][8 * hi]);
    pf.h[1] = *(const v8bf*)(&Pbuf[w * 16 + r][16 + 8 * hi]);
    for (int c = 0; c < 4; ++c) {
      V16U vf;
      vf.h[0] = *(const v8bf*)(&VtT[16 * c + r][8 * hi]);
      vf.h[1] = *(const v8bf*)(&VtT[16 * c + r][16 + 8 * hi]);
      O[c] = wmma_bf16(pf.v, vf.v, O[c]);
    }
  }

  // Normalize and write f32 head output
  float linv[8];
  for (int v = 0; v < 8; ++v) linv[v] = 1.f / lRun[w * 16 + v + 8 * hi];
  for (int c = 0; c < 4; ++c) {
    for (int v = 0; v < 8; ++v) {
      int row = rowBase + w * 16 + v + 8 * hi;
      attnO[((size_t)(b * N_ + row)) * DM + h * DH + 16 * c + r] = O[c][v] * linv[v];
    }
  }
}

// ---------------------------------------------------------------------------
extern "C" void kernel_launch(void* const* d_in, const int* in_sizes, int n_in,
                              void* d_out, int out_size, void* d_ws, size_t ws_size,
                              hipStream_t stream)
{
  (void)in_sizes; (void)n_in; (void)out_size; (void)ws_size;
  const float* x   = (const float*)d_in[0];
  const float* sim = (const float*)d_in[1];
  const float* Wq  = (const float*)d_in[2];
  const float* bq  = (const float*)d_in[3];
  const float* Wk  = (const float*)d_in[4];
  const float* bk  = (const float*)d_in[5];
  const float* Wv  = (const float*)d_in[6];
  const float* bv  = (const float*)d_in[7];
  const float* Wo  = (const float*)d_in[8];
  const float* bo  = (const float*)d_in[9];
  const float* tau = (const float*)d_in[10];

  const int M = B_ * N_;                       // 8192
  char* ws = (char*)d_ws;
  size_t off = 0;
  __bf16* Qb = (__bf16*)(ws + off); off += (size_t)M * DM * sizeof(__bf16);
  __bf16* Kb = (__bf16*)(ws + off); off += (size_t)M * DM * sizeof(__bf16);
  __bf16* Vb = (__bf16*)(ws + off); off += (size_t)M * DM * sizeof(__bf16);
  int*   biasIdx = (int*)(ws + off);   off += (size_t)M * TOPK_ * sizeof(int);
  float* biasVal = (float*)(ws + off); off += (size_t)M * TOPK_ * sizeof(float);
  float* attnO   = (float*)(ws + off); off += (size_t)M * DM * sizeof(float);
  float* outF = (float*)d_out;

  dim3 blk(256);
  dim3 gProj(M / 128, DM / 64);

  gemm_kernel<<<gProj, blk, 0, stream>>>(x, Wq, bq, Qb, nullptr, M, DIN, DM, 0);
  gemm_kernel<<<gProj, blk, 0, stream>>>(x, Wk, bk, Kb, nullptr, M, DIN, DM, 0);
  gemm_kernel<<<gProj, blk, 0, stream>>>(x, Wv, bv, Vb, nullptr, M, DIN, DM, 0);

  topk_kernel<<<dim3(M), blk, 0, stream>>>(sim, biasIdx, biasVal);

  attn_kernel<<<dim3(N_ / 128, B_ * H_), blk, 0, stream>>>(
      Qb, Kb, Vb, biasIdx, biasVal, tau, attnO);

  gemm_kernel<<<gProj, blk, 0, stream>>>(attnO, Wo, bo, nullptr, outF, M, DM, DM, 1);
}
